// SpatialLinkingModule_56745107915254
// MI455X (gfx1250) — matrix operand order repaired and checked by reading, not compiled, not run on hardware
//
#include <hip/hip_runtime.h>

// ---------------- problem constants ----------------
constexpr int kB  = 2;
constexpr int kS  = 4608;
constexpr int kD  = 3584;
constexpr int kNH = 8;
constexpr int kDH = 448;
constexpr int kP  = 4096;
constexpr int kNB = 12;
constexpr int kD2 = 7168;
constexpr int kM  = kNB * kNH;       // 96 attention rows per batch
constexpr float kEPS = 1e-5f;

typedef __attribute__((ext_vector_type(16))) __bf16 v16bf;
typedef __attribute__((ext_vector_type(8)))  float  v8f;

union Frag { uint4 q[2]; v16bf v; };

// ---------------- helpers ----------------
__device__ __forceinline__ unsigned short bf16rn(float f) {
  unsigned u = __float_as_uint(f);
  u += 0x7FFFu + ((u >> 16) & 1u);
  return (unsigned short)(u >> 16);
}
// split two f32 into packed bf16 hi and bf16 lo words
__device__ __forceinline__ void split2(float f0, float f1, unsigned& hi, unsigned& lo) {
  unsigned short h0 = bf16rn(f0), h1 = bf16rn(f1);
  float r0 = f0 - __uint_as_float(((unsigned)h0) << 16);
  float r1 = f1 - __uint_as_float(((unsigned)h1) << 16);
  unsigned short l0 = bf16rn(r0), l1 = bf16rn(r1);
  hi = (unsigned)h0 | ((unsigned)h1 << 16);
  lo = (unsigned)l0 | ((unsigned)l1 << 16);
}

// 3x bf16 WMMA emulating fp32 GEMM: accA += ah*bh ; accB += ah*bl + al*bh
__device__ __forceinline__ void wmma3(v8f& accA, v8f& accB,
                                      const unsigned* sAh, const unsigned* sAl,
                                      const unsigned* sBh, const unsigned* sBl,
                                      int rowA17, int rowB17, int kg, int hl) {
  const uint4* pAh = (const uint4*)sAh;
  const uint4* pAl = (const uint4*)sAl;
  const uint4* pBh = (const uint4*)sBh;
  const uint4* pBl = (const uint4*)sBl;
  int a0 = rowA17 + 4 * kg + hl;       // A frag: K = 8h..8h+7 of this 32-group
  int a1 = a0 + 2;                     //         K = 16+8h..16+8h+7
  int b0 = rowB17 + 4 * kg + hl;       // B frag: K = 8h..8h+7, then 16+8h..16+8h+7
  Frag ah, al, bh, bl;
  ah.q[0] = pAh[a0]; ah.q[1] = pAh[a1];
  al.q[0] = pAl[a0]; al.q[1] = pAl[a1];
  bh.q[0] = pBh[b0]; bh.q[1] = pBh[b0 + 2];
  bl.q[0] = pBl[b0]; bl.q[1] = pBl[b0 + 2];
  accA = __builtin_amdgcn_wmma_f32_16x16x32_bf16(false, ah.v, false, bh.v, (short)0, accA, false, false);
  accB = __builtin_amdgcn_wmma_f32_16x16x32_bf16(false, ah.v, false, bl.v, (short)0, accB, false, false);
  accB = __builtin_amdgcn_wmma_f32_16x16x32_bf16(false, al.v, false, bh.v, (short)0, accB, false, false);
}

// ---------------- kernel 0: copy embeds -> out ----------------
__global__ __launch_bounds__(256) void k_copy(const float4* __restrict__ src,
                                              float4* __restrict__ dst, int n4) {
  int i = blockIdx.x * 256 + threadIdx.x;
  if (i < n4) dst[i] = src[i];
}

// ---------------- kernel 1: find box-end positions + image start ----------------
__global__ __launch_bounds__(256) void k_find(const int* __restrict__ ids,
                                              const unsigned char* __restrict__ imask,
                                              const int* __restrict__ boxEndId,
                                              int* __restrict__ pos, int* __restrict__ start) {
  int b = blockIdx.x;
  __shared__ int cnt, st, tmp[64];
  if (threadIdx.x == 0) { cnt = 0; st = kS; }
  __syncthreads();
  int be = boxEndId[0];
  for (int s = threadIdx.x; s < kS; s += 256) {
    if (ids[b * kS + s] == be) {
      int k = atomicAdd(&cnt, 1);
      if (k < 64) tmp[k] = s;
    }
    if (imask[b * kS + s]) atomicMin(&st, s);
  }
  __syncthreads();
  if (threadIdx.x == 0) {
    int m = cnt < 64 ? cnt : 64;
    for (int i = 1; i < m; ++i) {           // insertion sort (tiny)
      int key = tmp[i], j = i - 1;
      while (j >= 0 && tmp[j] > key) { tmp[j + 1] = tmp[j]; --j; }
      tmp[j + 1] = key;
    }
    for (int i = 0; i < kNB; ++i) pos[b * kNB + i] = (i < m) ? tmp[i] : 0;
    start[b] = st;
  }
}

// ---------------- kernel 2: gather q_tok ----------------
__global__ __launch_bounds__(256) void k_gather(const float* __restrict__ E,
                                                const int* __restrict__ pos,
                                                float* __restrict__ qtok) {
  int idx = blockIdx.x * 256 + threadIdx.x;          // < 24*3584
  int r = idx / kD, d = idx - r * kD;
  int b = r / kNB;
  qtok[idx] = E[((size_t)b * kS + pos[r]) * kD + d];
}

// ---------------- skinny GEMMs: one wave per output column, 24 rows ----------------
#define REDUCE24(acc)                                            \
  _Pragma("unroll") for (int r = 0; r < 24; ++r)                 \
    for (int o = 16; o; o >>= 1) acc[r] += __shfl_xor(acc[r], o);

// q = (q_tok @ Wq^T + bq) * scale     Wq = in_proj rows [0,D)
__global__ __launch_bounds__(256) void k_qproj(const float* __restrict__ X,
                                               const float* __restrict__ W,
                                               const float* __restrict__ bias,
                                               float* __restrict__ outq) {
  int gw = (blockIdx.x * 256 + threadIdx.x) >> 5;
  int lane = threadIdx.x & 31;
  if (gw >= kD) return;
  float acc[24];
#pragma unroll
  for (int r = 0; r < 24; ++r) acc[r] = 0.f;
  const float4* wp = (const float4*)(W + (size_t)gw * kD);
  for (int it = 0; it < kD / 128; ++it) {
    float4 w4 = wp[it * 32 + lane];
#pragma unroll
    for (int r = 0; r < 24; ++r) {
      float4 x4 = ((const float4*)(X + (size_t)r * kD))[it * 32 + lane];
      acc[r] = fmaf(w4.x, x4.x, fmaf(w4.y, x4.y, fmaf(w4.z, x4.z, fmaf(w4.w, x4.w, acc[r]))));
    }
  }
  REDUCE24(acc)
  if (lane == 0) {
    float scl = 1.0f / sqrtf((float)kDH);
#pragma unroll
    for (int r = 0; r < 24; ++r) outq[(size_t)r * kD + gw] = (acc[r] + bias[gw]) * scl;
  }
}

// qk[r,h,j] = sum_i q[r, h*448+i] * Wk[h*448+i, j]   (weight-stationary over j)
__global__ __launch_bounds__(256) void k_qk(const float* __restrict__ q,
                                            const float* __restrict__ inW,
                                            float* __restrict__ qk) {
  __shared__ float qs[24 * kDH];
  int h = blockIdx.y;
  int j = blockIdx.x * 256 + threadIdx.x;
  for (int u = threadIdx.x; u < 24 * kDH; u += 256) {
    int r = u / kDH, i = u - r * kDH;
    qs[u] = q[(size_t)r * kD + h * kDH + i];
  }
  __syncthreads();
  float acc[24];
#pragma unroll
  for (int r = 0; r < 24; ++r) acc[r] = 0.f;
  const float* wcol = inW + ((size_t)kD + h * kDH) * kD + j;   // Wk = rows [D,2D)
  for (int i = 0; i < kDH; ++i) {
    float wv = wcol[(size_t)i * kD];
#pragma unroll
    for (int r = 0; r < 24; ++r) acc[r] = fmaf(qs[r * kDH + i], wv, acc[r]);
  }
#pragma unroll
  for (int r = 0; r < 24; ++r) qk[(size_t)(r * kNH + h) * kD + j] = acc[r];
}

// ---------------- WMMA GEMM 1: scores[b,m,p] = qk[b,m,:] . patches[b,p,:] ----------------
__global__ __launch_bounds__(192) void k_scores(const float* __restrict__ qk,
                                                const float* __restrict__ E,
                                                const int* __restrict__ startArr,
                                                float* __restrict__ sc) {
  __shared__ __align__(16) unsigned sAh[96 * 68], sAl[96 * 68], sBh[16 * 68], sBl[16 * 68];
  int b = blockIdx.y, p0 = blockIdx.x * 16;
  int st = startArr[b];
  int tid = threadIdx.x, w = tid >> 5, lane = tid & 31;
  int ml = lane & 15, hl = lane >> 4;
  int rowA17 = (w * 16 + ml) * 17;
  int rowB17 = ml * 17;
  const float* Abase = qk + (size_t)(b * kM) * kD;
  const float* Bbase = E + ((size_t)b * kS + st + p0) * kD;
  v8f accA = {}, accB = {};
  for (int k0 = 0; k0 < kD; k0 += 128) {
    __syncthreads();
#pragma unroll 4
    for (int it = 0; it < 32; ++it) {                   // A: 96x128 -> split bf16
      int u = tid + 192 * it;
      int m = u >> 6, kp = u & 63;
      float2 f2 = *(const float2*)(Abase + (size_t)m * kD + k0 + 2 * kp);
      unsigned hi, lo; split2(f2.x, f2.y, hi, lo);
      sAh[m * 68 + kp] = hi; sAl[m * 68 + kp] = lo;
    }
    for (int u = tid; u < 1024; u += 192) {             // B: 16 patches x 128 (row-major)
      int n = u >> 6, kp = u & 63;
      float2 f2 = *(const float2*)(Bbase + (size_t)n * kD + k0 + 2 * kp);
      unsigned hi, lo; split2(f2.x, f2.y, hi, lo);
      sBh[n * 68 + kp] = hi; sBl[n * 68 + kp] = lo;
    }
    if (k0 + 128 < kD) {                                // pull next chunk toward L2
      __builtin_prefetch(Abase + (size_t)(tid % 96) * kD + k0 + 128, 0, 1);
      __builtin_prefetch(Bbase + (size_t)(tid & 15) * kD + k0 + 128, 0, 1);
    }
    __syncthreads();
#pragma unroll
    for (int kg = 0; kg < 4; ++kg)
      wmma3(accA, accB, sAh, sAl, sBh, sBl, rowA17, rowB17, kg, hl);
  }
  float* out = sc + (size_t)(b * kM) * kP + p0;
#pragma unroll
  for (int r = 0; r < 8; ++r) {
    int m = w * 16 + hl * 8 + r;
    out[(size_t)m * kP + ml] = accA[r] + accB[r];
  }
}

// ---------------- masked softmax over patches, in place ----------------
__global__ __launch_bounds__(256) void k_softmax(float* __restrict__ sc,
                                                 const float* __restrict__ boxes,
                                                 const int* __restrict__ thw) {
  int bid = blockIdx.x;            // (b*12+n)*8+h
  int bn = bid >> 3;
  int b = bn / kNB, n = bn - b * kNB;
  int h = thw[1], w = thw[2];
  const float* bx = boxes + (size_t)(b * kNB + n) * 4;
  int px1 = min(max((int)floorf(bx[0] * w), 0), w - 1);
  int px2 = max(px1 + 1, min((int)floorf(bx[2] * w), w));
  int py1 = min(max((int)floorf(bx[1] * h), 0), h - 1);
  int py2 = max(py1 + 1, min((int)floorf(bx[3] * h), h));
  int hw = h * w;
  float* row = sc + (size_t)bid * kP;
  int tid = threadIdx.x;
  __shared__ float red[8];
  __shared__ float sMax, sSum;
  float mx = -3.0e38f;
  for (int p = tid; p < kP; p += 256) {
    int q = p % hw; int py = q / w; int px = q - py * w;
    bool in = (py >= py1) && (py < py2) && (px >= px1) && (px < px2);
    float v = in ? row[p] : -1e30f;
    mx = fmaxf(mx, v);
  }
  for (int o = 16; o; o >>= 1) mx = fmaxf(mx, __shfl_xor(mx, o));
  if ((tid & 31) == 0) red[tid >> 5] = mx;
  __syncthreads();
  if (tid == 0) { float v = red[0]; for (int i = 1; i < 8; ++i) v = fmaxf(v, red[i]); sMax = v; }
  __syncthreads();
  mx = sMax;
  float s = 0.f;
  for (int p = tid; p < kP; p += 256) {
    int q = p % hw; int py = q / w; int px = q - py * w;
    bool in = (py >= py1) && (py < py2) && (px >= px1) && (px < px2);
    float e = in ? expf(row[p] - mx) : 0.f;
    row[p] = e;
    s += e;
  }
  for (int o = 16; o; o >>= 1) s += __shfl_xor(s, o);
  if ((tid & 31) == 0) red[tid >> 5] = s;
  __syncthreads();
  if (tid == 0) { float v = 0.f; for (int i = 0; i < 8; ++i) v += red[i]; sSum = v; }
  __syncthreads();
  float inv = 1.0f / sSum;
  for (int p = tid; p < kP; p += 256) row[p] *= inv;
}

// ---------------- WMMA GEMM 2: ctxP[b,m,j] = sum_p attn[b,m,p] * patches[b,p,j] ----------------
__global__ __launch_bounds__(192) void k_ctxp(const float* __restrict__ attn,
                                              const float* __restrict__ E,
                                              const int* __restrict__ startArr,
                                              float* __restrict__ ctxP) {
  __shared__ __align__(16) unsigned sAh[96 * 68], sAl[96 * 68], sBh[16 * 68], sBl[16 * 68];
  int b = blockIdx.y, n0 = blockIdx.x * 16;
  int st = startArr[b];
  int tid = threadIdx.x, w = tid >> 5, lane = tid & 31;
  int ml = lane & 15, hl = lane >> 4;
  int rowA17 = (w * 16 + ml) * 17;
  int rowB17 = ml * 17;
  const float* Abase = attn + (size_t)(b * kM) * kP;
  const float* Bbase = E + ((size_t)b * kS + st) * kD + n0;
  v8f accA = {}, accB = {};
  for (int k0 = 0; k0 < kP; k0 += 128) {
    __syncthreads();
#pragma unroll 4
    for (int it = 0; it < 32; ++it) {                   // A: attn 96x128
      int u = tid + 192 * it;
      int m = u >> 6, kp = u & 63;
      float2 f2 = *(const float2*)(Abase + (size_t)m * kP + k0 + 2 * kp);
      unsigned hi, lo; split2(f2.x, f2.y, hi, lo);
      sAh[m * 68 + kp] = hi; sAl[m * 68 + kp] = lo;
    }
    for (int u = tid; u < 1024; u += 192) {             // B: 128 patch-rows x 16 cols -> LDS [n][k]
      int n = u & 15, kp = u >> 4;
      float f0 = Bbase[(size_t)(k0 + 2 * kp) * kD + n];
      float f1 = Bbase[(size_t)(k0 + 2 * kp + 1) * kD + n];
      unsigned hi, lo; split2(f0, f1, hi, lo);
      sBh[n * 68 + kp] = hi; sBl[n * 68 + kp] = lo;
    }
    if (k0 + 128 < kP)
      __builtin_prefetch(Bbase + (size_t)(k0 + 128 + (tid >> 1)) * kD, 0, 1);
    __syncthreads();
#pragma unroll
    for (int kg = 0; kg < 4; ++kg)
      wmma3(accA, accB, sAh, sAl, sBh, sBl, rowA17, rowB17, kg, hl);
  }
  float* out = ctxP + (size_t)(b * kM) * kD + n0;
#pragma unroll
  for (int r = 0; r < 8; ++r) {
    int m = w * 16 + hl * 8 + r;
    out[(size_t)m * kD + ml] = accA[r] + accB[r];
  }
}

// ctx[r,c] = sum_j ctxP[r, h(c), j] * Wv[c, j] + bv[c]   (Wv = in_proj rows [2D,3D))
__global__ __launch_bounds__(256) void k_ctxd(const float* __restrict__ ctxP,
                                              const float* __restrict__ inW,
                                              const float* __restrict__ inB,
                                              float* __restrict__ ctx) {
  int gw = (blockIdx.x * 256 + threadIdx.x) >> 5;
  int lane = threadIdx.x & 31;
  if (gw >= kD) return;
  int h = gw / kDH;
  float acc[24];
#pragma unroll
  for (int r = 0; r < 24; ++r) acc[r] = 0.f;
  const float4* wp = (const float4*)(inW + ((size_t)2 * kD + gw) * kD);
  for (int it = 0; it < kD / 128; ++it) {
    float4 w4 = wp[it * 32 + lane];
#pragma unroll
    for (int r = 0; r < 24; ++r) {
      float4 x4 = ((const float4*)(ctxP + (size_t)(r * kNH + h) * kD))[it * 32 + lane];
      acc[r] = fmaf(w4.x, x4.x, fmaf(w4.y, x4.y, fmaf(w4.z, x4.z, fmaf(w4.w, x4.w, acc[r]))));
    }
  }
  REDUCE24(acc)
  if (lane == 0) {
    float bv = inB[2 * kD + gw];
#pragma unroll
    for (int r = 0; r < 24; ++r) ctx[(size_t)r * kD + gw] = acc[r] + bv;
  }
}

// comb[r,c] = q_tok[r,c] + ctx[r,:]·Wo[c,:] + bo[c]
__global__ __launch_bounds__(256) void k_outproj(const float* __restrict__ ctx,
                                                 const float* __restrict__ qtok,
                                                 const float* __restrict__ W,
                                                 const float* __restrict__ bias,
                                                 float* __restrict__ comb) {
  int gw = (blockIdx.x * 256 + threadIdx.x) >> 5;
  int lane = threadIdx.x & 31;
  if (gw >= kD) return;
  float acc[24];
#pragma unroll
  for (int r = 0; r < 24; ++r) acc[r] = 0.f;
  const float4* wp = (const float4*)(W + (size_t)gw * kD);
  for (int it = 0; it < kD / 128; ++it) {
    float4 w4 = wp[it * 32 + lane];
#pragma unroll
    for (int r = 0; r < 24; ++r) {
      float4 x4 = ((const float4*)(ctx + (size_t)r * kD))[it * 32 + lane];
      acc[r] = fmaf(w4.x, x4.x, fmaf(w4.y, x4.y, fmaf(w4.z, x4.z, fmaf(w4.w, x4.w, acc[r]))));
    }
  }
  REDUCE24(acc)
  if (lane == 0) {
#pragma unroll
    for (int r = 0; r < 24; ++r)
      comb[(size_t)r * kD + gw] = qtok[(size_t)r * kD + gw] + acc[r] + bias[gw];
  }
}

// LayerNorm in place on comb rows
__global__ __launch_bounds__(256) void k_ln(float* __restrict__ comb,
                                            const float* __restrict__ g,
                                            const float* __restrict__ be) {
  float* row = comb + (size_t)blockIdx.x * kD;
  int tid = threadIdx.x;
  float s = 0.f, s2 = 0.f;
  for (int i = tid; i < kD; i += 256) { float v = row[i]; s += v; s2 += v * v; }
  __shared__ float rs[8], rs2[8];
  __shared__ float sMu, sR;
  for (int o = 16; o; o >>= 1) { s += __shfl_xor(s, o); s2 += __shfl_xor(s2, o); }
  if ((tid & 31) == 0) { rs[tid >> 5] = s; rs2[tid >> 5] = s2; }
  __syncthreads();
  if (tid == 0) {
    float t = 0.f, t2 = 0.f;
    for (int i = 0; i < 8; ++i) { t += rs[i]; t2 += rs2[i]; }
    float mu = t / kD;
    float var = t2 / kD - mu * mu;
    sMu = mu; sR = rsqrtf(var + kEPS);
  }
  __syncthreads();
  float mu = sMu, rstd = sR;
  for (int i = tid; i < kD; i += 256)
    row[i] = (row[i] - mu) * rstd * g[i] + be[i];
}

// hidden[r,c] = gelu_exact(comb[r,:]·W1[c,:] + b1[c])
__global__ __launch_bounds__(256) void k_mlp1(const float* __restrict__ comb,
                                              const float* __restrict__ W1,
                                              const float* __restrict__ b1,
                                              float* __restrict__ hidden) {
  int gw = (blockIdx.x * 256 + threadIdx.x) >> 5;
  int lane = threadIdx.x & 31;
  if (gw >= kD2) return;
  float acc[24];
#pragma unroll
  for (int r = 0; r < 24; ++r) acc[r] = 0.f;
  const float4* wp = (const float4*)(W1 + (size_t)gw * kD);
  for (int it = 0; it < kD / 128; ++it) {
    float4 w4 = wp[it * 32 + lane];
#pragma unroll
    for (int r = 0; r < 24; ++r) {
      float4 x4 = ((const float4*)(comb + (size_t)r * kD))[it * 32 + lane];
      acc[r] = fmaf(w4.x, x4.x, fmaf(w4.y, x4.y, fmaf(w4.z, x4.z, fmaf(w4.w, x4.w, acc[r]))));
    }
  }
  REDUCE24(acc)
  if (lane == 0) {
#pragma unroll
    for (int r = 0; r < 24; ++r) {
      float x = acc[r] + b1[gw];
      hidden[(size_t)r * kD2 + gw] = 0.5f * x * (1.0f + erff(x * 0.70710678118654752f));
    }
  }
}

// out[b, pos[r], c] = comb[r,c] + hidden[r,:]·W2[c,:] + b2[c]
__global__ __launch_bounds__(256) void k_mlp2(const float* __restrict__ hidden,
                                              const float* __restrict__ comb,
                                              const float* __restrict__ W2,
                                              const float* __restrict__ b2,
                                              const int* __restrict__ pos,
                                              float* __restrict__ out) {
  int gw = (blockIdx.x * 256 + threadIdx.x) >> 5;
  int lane = threadIdx.x & 31;
  if (gw >= kD) return;
  float acc[24];
#pragma unroll
  for (int r = 0; r < 24; ++r) acc[r] = 0.f;
  const float4* wp = (const float4*)(W2 + (size_t)gw * kD2);
  for (int it = 0; it < kD2 / 128; ++it) {
    float4 w4 = wp[it * 32 + lane];
#pragma unroll
    for (int r = 0; r < 24; ++r) {
      float4 x4 = ((const float4*)(hidden + (size_t)r * kD2))[it * 32 + lane];
      acc[r] = fmaf(w4.x, x4.x, fmaf(w4.y, x4.y, fmaf(w4.z, x4.z, fmaf(w4.w, x4.w, acc[r]))));
    }
  }
  REDUCE24(acc)
  if (lane == 0) {
#pragma unroll
    for (int r = 0; r < 24; ++r) {
      int b = r / kNB;
      out[((size_t)b * kS + pos[r]) * kD + gw] = comb[(size_t)r * kD + gw] + acc[r] + b2[gw];
    }
  }
}

// ---------------- host launcher ----------------
extern "C" void kernel_launch(void* const* d_in, const int* in_sizes, int n_in,
                              void* d_out, int out_size, void* d_ws, size_t ws_size,
                              hipStream_t stream) {
  const float* E      = (const float*)d_in[0];
  const int*   ids    = (const int*)d_in[1];
  const unsigned char* imask = (const unsigned char*)d_in[2];
  const float* boxes  = (const float*)d_in[3];
  const int*   thw    = (const int*)d_in[4];
  const int*   boxEnd = (const int*)d_in[5];
  const float* inW    = (const float*)d_in[6];
  const float* inB    = (const float*)d_in[7];
  const float* Wo     = (const float*)d_in[8];
  const float* bo     = (const float*)d_in[9];
  const float* gam    = (const float*)d_in[10];
  const float* bet    = (const float*)d_in[11];
  const float* W1     = (const float*)d_in[12];
  const float* b1     = (const float*)d_in[13];
  const float* W2     = (const float*)d_in[14];
  const float* b2     = (const float*)d_in[15];
  float* out = (float*)d_out;

  char* w = (char*)d_ws;
  int*   pos   = (int*)w;                       // 24 ints
  int*   start = (int*)(w + 96);                // 2 ints
  float* qtok  = (float*)(w + 256);             // [24, D]
  float* q     = qtok + (size_t)24 * kD;        // [24, D]
  float* qk    = q + (size_t)24 * kD;           // [192, D]
  float* attn  = qk + (size_t)192 * kD;         // [192, P]  (scores -> attn in place)
  float* ctxP  = attn + (size_t)192 * kP;       // [192, D]
  float* ctx   = ctxP + (size_t)192 * kD;       // [24, D]
  float* comb  = ctx + (size_t)24 * kD;         // [24, D]
  float* hidden= comb + (size_t)24 * kD;        // [24, 2D]

  int n4 = (kB * kS * kD) / 4;
  k_copy   <<<(n4 + 255) / 256, 256, 0, stream>>>((const float4*)E, (float4*)out, n4);
  k_find   <<<kB, 256, 0, stream>>>(ids, imask, boxEnd, pos, start);
  k_gather <<<(24 * kD) / 256, 256, 0, stream>>>(E, pos, qtok);
  k_qproj  <<<(kD * 32) / 256, 256, 0, stream>>>(qtok, inW, inB, q);
  k_qk     <<<dim3(kD / 256, kNH), 256, 0, stream>>>(q, inW, qk);
  k_scores <<<dim3(kP / 16, kB), 192, 0, stream>>>(qk, E, start, attn);
  k_softmax<<<kB * kNB * kNH, 256, 0, stream>>>(attn, boxes, thw);
  k_ctxp   <<<dim3(kD / 16, kB), 192, 0, stream>>>(attn, E, start, ctxP);
  k_ctxd   <<<(kD * 32) / 256, 256, 0, stream>>>(ctxP, inW, inB, ctx);
  k_outproj<<<(kD * 32) / 256, 256, 0, stream>>>(ctx, qtok, Wo, bo, comb);
  k_ln     <<<24, 256, 0, stream>>>(comb, gam, bet);
  k_mlp1   <<<(kD2 * 32) / 256, 256, 0, stream>>>(comb, W1, b1, hidden);
  k_mlp2   <<<(kD * 32) / 256, 256, 0, stream>>>(hidden, comb, W2, b2, pos, out);
}
